// HungarianMatcher_70282844832272
// MI455X (gfx1250) — compile-verified, matrix operand order checked by reading, not run on hardware
//
#include <hip/hip_runtime.h>
#include <hip/hip_bf16.h>

// ---------------- problem constants (match reference) ----------------
constexpr int Bc = 8, Pc = 16384, Qc = 256, Ec = 128;
constexpr int QB = 4;           // Q blocks of 64
constexpr int KS = 8;           // P split into 8 segments of 2048
constexpr int KITER = Pc / KS / 32;   // 64 K-steps of 32 per block

// cost weights
#define C_CLASS 2.0f
#define C_MASK  5.0f
#define C_DICE  5.0f
#define C_DIST  1.0f
#define C_NLL   0.5f
#define C_LIK   0.5f

// ---------------- workspace layout (floats) ----------------
constexpr long OFF_RMAX = 0;                       // B*P
constexpr long OFF_RSUM = OFF_RMAX + (long)Bc*Pc;  // B*P
constexpr long OFF_ACCP = OFF_RSUM + (long)Bc*Pc;  // B*Q*E  (zeroed)
constexpr long OFF_ACCN = OFF_ACCP + (long)Bc*Qc*Ec;
constexpr long OFF_ACCM = OFF_ACCN + (long)Bc*Qc*Ec;
constexpr long OFF_NEGC = OFF_ACCM + (long)Bc*Qc*Ec;  // B*Q
constexpr long OFF_PPC  = OFF_NEGC + (long)Bc*Qc;     // B*Q
constexpr long OFF_SEGC = OFF_PPC  + (long)Bc*Qc;     // B*E
constexpr long OFF_NNZ  = OFF_SEGC + (long)Bc*Ec;     // B
constexpr long WS_TOTAL = OFF_NNZ + Bc;
constexpr long ZERO_OFF = OFF_ACCP;
constexpr long ZERO_LEN = WS_TOTAL - OFF_ACCP;

// ---------------- CDNA5 WMMA types ----------------
typedef __attribute__((ext_vector_type(16))) __bf16 v16bf;
typedef __attribute__((ext_vector_type(8)))  float  v8f;

__device__ __forceinline__ float splus(float x) {
    return fmaxf(x, 0.0f) + log1pf(__expf(-fabsf(x)));
}
__device__ __forceinline__ unsigned short f2bf(float f) {
    unsigned int u = __float_as_uint(f);
    unsigned int r = u + 0x7fffu + ((u >> 16) & 1u);   // round-to-nearest-even
    return (unsigned short)(r >> 16);
}
// Load one 16x16x32 bf16 A/B fragment row from LDS.
// Element j of the per-lane fragment holds K = half*8 + (j<8 ? 0 : 16) + (j&7)
// -> two contiguous 8-half (16B) runs -> two ds_load_b128.
__device__ __forceinline__ v16bf load_frag(const unsigned short* row, int half) {
    union { uint4 u[2]; v16bf v; } t;
    const unsigned short* p = row + half * 8;
    t.u[0] = *reinterpret_cast<const uint4*>(p);
    t.u[1] = *reinterpret_cast<const uint4*>(p + 16);
    return t.v;
}

// ================= K1: masked row max + sum(exp) over Q, wave per row ==========
__global__ __launch_bounds__(256)
void k_rowstats(const float* __restrict__ lgv, const unsigned char* __restrict__ msk,
                float* __restrict__ rmax, float* __restrict__ rsum) {
    int r    = blockIdx.x * 8 + (threadIdx.x >> 5);   // B*P rows
    int lane = threadIdx.x & 31;
    long base = (long)r * Qc;
    float xs[8]; bool ms[8];
    float mx = -1e30f;
    #pragma unroll
    for (int i = 0; i < 8; ++i) {
        int q = lane + i * 32;
        float x = lgv[base + q];
        bool  m = msk[base + q] != 0;
        xs[i] = x; ms[i] = m;
        mx = fmaxf(mx, m ? x : -1e30f);
    }
    #pragma unroll
    for (int o = 16; o > 0; o >>= 1) mx = fmaxf(mx, __shfl_xor(mx, o, 32));
    float s = 0.0f;
    #pragma unroll
    for (int i = 0; i < 8; ++i) s += ms[i] ? __expf(xs[i] - mx) : 0.0f;
    #pragma unroll
    for (int o = 16; o > 0; o >>= 1) s += __shfl_xor(s, o, 32);
    if (lane == 0) { rmax[r] = mx; rsum[r] = s; }
}

// ================= K2: segmap column sums + nnz ================================
__global__ __launch_bounds__(128)
void k_segstats(const float* __restrict__ segv, float* __restrict__ segc,
                float* __restrict__ nnz) {
    int b  = blockIdx.x >> 4;
    int ks = blockIdx.x & 15;
    int e  = threadIdx.x;          // 128 threads = full E, coalesced
    float s = 0.0f, c = 0.0f;
    int p0 = ks * (Pc / 16);
    for (int p = p0; p < p0 + Pc / 16; ++p) {
        float v = segv[(long)(b * Pc + p) * Ec + e];
        s += v;
        c += (v > 0.0f) ? 1.0f : 0.0f;
    }
    unsafeAtomicAdd(&segc[b * Ec + e], s);
    unsafeAtomicAdd(&nnz[b], c);
}

// ================= K3: fused triple GEMM over P with bf16 WMMA =================
// block = (image b, 64-query block qb, P segment ks); 256 threads = 8 waves.
// wave w: Q-tile = w>>1 (16 q), E-half = w&1 (4 E-tiles of 16) -> 12 v8f accums.
__global__ __launch_bounds__(256)
void k_gemm3(const float* __restrict__ lgv, const unsigned char* __restrict__ msk,
             const float* __restrict__ segv,
             const float* __restrict__ rmax, const float* __restrict__ rsum,
             float* __restrict__ accP, float* __restrict__ accN, float* __restrict__ accM,
             float* __restrict__ negc, float* __restrict__ ppc) {
    __shared__ __align__(16) unsigned short As[3][64][40];   // [pos|neg|predp][q][k]
    __shared__ __align__(16) unsigned short Bs[2][128][40];  // [targ|segv][e][k]
    __shared__ float colNeg[64], colPred[64];

    const int tid = threadIdx.x;
    const int b   = blockIdx.x / (QB * KS);
    const int rem = blockIdx.x % (QB * KS);
    const int qb  = rem / KS;
    const int ks  = rem % KS;
    const int pbase = ks * (Pc / KS);

    if (tid < 64) { colNeg[tid] = 0.0f; colPred[tid] = 0.0f; }

    const int wave = tid >> 5, lane = tid & 31;
    const int half = lane >> 4, l16 = lane & 15;
    const int qtile = wave >> 1, ehalf = wave & 1;

    // staging decomposition
    const int qloc = tid & 63, prow = tid >> 6;   // A: 64q x 32p, 8 elems/thread
    const int eloc = tid & 127, pgrp = tid >> 7;  // B: 128e x 32p, 16 elems/thread x2

    v8f zf; for (int k = 0; k < 8; ++k) zf[k] = 0.0f;
    v8f aP[4], aN[4], aM[4];
    #pragma unroll
    for (int i = 0; i < 4; ++i) { aP[i] = zf; aN[i] = zf; aM[i] = zf; }

    for (int it = 0; it < KITER; ++it) {
        __syncthreads();            // protect LDS reuse (also covers colsum init)
        const int p0 = pbase + it * 32;
        // ---- stage A tiles: compute pos / neg / pred_p in f32, store bf16 ----
        #pragma unroll
        for (int i = 0; i < 8; ++i) {
            int p = i * 4 + prow;
            long row = (long)(b * Pc + p0 + p);
            float rm = rmax[row], rs = rsum[row];
            long g = row * Qc + qb * 64 + qloc;
            float x = lgv[g];
            bool  m = msk[g] != 0;
            float pos = m ? splus(-x) : 0.0f;
            float neg = m ? splus(x)  : 0.0f;
            float ez  = m ? __expf(x - rm) : 0.0f;
            float pp  = (rs > 0.0f) ? ez / fmaxf(rs, 1e-30f) : 0.0f;
            As[0][qloc][p] = f2bf(pos);
            As[1][qloc][p] = f2bf(neg);
            As[2][qloc][p] = f2bf(pp);
            atomicAdd(&colNeg[qloc], neg);    // ds_add_f32
            atomicAdd(&colPred[qloc], pp);
        }
        // ---- stage B tiles: targ (binarized) and segmap values ----
        #pragma unroll
        for (int i = 0; i < 16; ++i) {
            int p = pgrp * 16 + i;
            float v = segv[(long)(b * Pc + p0 + p) * Ec + eloc];
            Bs[0][eloc][p] = f2bf(v > 0.0f ? 1.0f : 0.0f);
            Bs[1][eloc][p] = f2bf(v);
        }
        __syncthreads();
        // ---- WMMA: D = A(16q x 32p) * B(32p x 16e) + C ----
        v16bf fP = load_frag(&As[0][qtile * 16 + l16][0], half);
        v16bf fN = load_frag(&As[1][qtile * 16 + l16][0], half);
        v16bf fM = load_frag(&As[2][qtile * 16 + l16][0], half);
        #pragma unroll
        for (int et = 0; et < 4; ++et) {
            int e = ehalf * 64 + et * 16 + l16;
            v16bf bT = load_frag(&Bs[0][e][0], half);
            v16bf bV = load_frag(&Bs[1][e][0], half);
            aP[et] = __builtin_amdgcn_wmma_f32_16x16x32_bf16(false, fP, false, bT, (short)0, aP[et], false, false);
            aN[et] = __builtin_amdgcn_wmma_f32_16x16x32_bf16(false, fN, false, bT, (short)0, aN[et], false, false);
            aM[et] = __builtin_amdgcn_wmma_f32_16x16x32_bf16(false, fM, false, bV, (short)0, aM[et], false, false);
        }
    }
    __syncthreads();
    if (tid < 64) {
        unsafeAtomicAdd(&negc[b * Qc + qb * 64 + tid], colNeg[tid]);
        unsafeAtomicAdd(&ppc [b * Qc + qb * 64 + tid], colPred[tid]);
    }
    // C/D layout: VGPR r -> M = r + half*8, N = l16
    #pragma unroll
    for (int et = 0; et < 4; ++et) {
        int e = ehalf * 64 + et * 16 + l16;
        #pragma unroll
        for (int r = 0; r < 8; ++r) {
            int q = qb * 64 + qtile * 16 + r + half * 8;
            long o = (long)(b * Qc + q) * Ec + e;
            unsafeAtomicAdd(&accP[o], aP[et][r]);
            unsafeAtomicAdd(&accN[o], aN[et][r]);
            unsafeAtomicAdd(&accM[o], aM[et][r]);
        }
    }
}

// ================= K4: assemble total cost [B,Q,E] =============================
__global__ __launch_bounds__(256)
void k_combine(const float* __restrict__ plog, const float* __restrict__ ppos,
               const float* __restrict__ chol, const float* __restrict__ tpos,
               const float* __restrict__ isz,
               const float* __restrict__ accP, const float* __restrict__ accN,
               const float* __restrict__ accM,
               const float* __restrict__ negc, const float* __restrict__ ppc,
               const float* __restrict__ segc, const float* __restrict__ nnz,
               float* __restrict__ out) {
    long idx = (long)blockIdx.x * blockDim.x + threadIdx.x;
    if (idx >= (long)Bc * Qc * Ec) return;
    int e = (int)(idx % Ec);
    int q = (int)((idx / Ec) % Qc);
    int b = (int)(idx / ((long)Qc * Ec));

    float nz = fmaxf(nnz[b], 1.0f);
    float mask_cost = (accP[idx] + (negc[b * Qc + q] - accN[idx])) / nz;
    float num  = 2.0f * accM[idx];
    float den  = ppc[b * Qc + q] + segc[b * Ec + e];
    float dice = 1.0f - (num + 1.0f) / (den + 1.0f);
    float cls  = splus(-plog[b * Qc + q]);

    float px = ppos[(b * Qc + q) * 2 + 0], py = ppos[(b * Qc + q) * 2 + 1];
    float tx = tpos[(b * Ec + e) * 2 + 0], ty = tpos[(b * Ec + e) * 2 + 1];
    float dx = px - tx, dy = py - ty;
    float ax = fabsf(dx), ay = fabsf(dy);
    float hub = 0.5f * ((ax < 1.0f ? 0.5f * dx * dx : ax - 0.5f) +
                        (ay < 1.0f ? 0.5f * dy * dy : ay - 0.5f));

    float sx = isz[b * 2 + 0], sy = isz[b * 2 + 1];
    const float* L = &chol[(long)(b * Qc + q) * 4];
    float L00 = L[0], L10 = L[2], L11 = L[3];
    float d0 = ty * sy - py * sy;   // diff components (y - mu)
    float d1 = d0;                  // placeholder, fixed below
    d0 = tx * sx - px * sx;
    d1 = ty * sy - py * sy;
    float z0 = d0 / L00;
    float z1 = (d1 - L10 * z0) / L11;
    float nll = 0.5f * (z0 * z0 + z1 * z1) + 1.8378770664093453f + __logf(L00) + __logf(L11);
    float lik = 1.0f - __expf(-nll);

    out[idx] = C_CLASS * cls + C_MASK * mask_cost + C_DICE * dice +
               C_DIST * hub + C_NLL * nll + C_LIK * lik;
}

// ================= host launcher ==============================================
extern "C" void kernel_launch(void* const* d_in, const int* in_sizes, int n_in,
                              void* d_out, int out_size, void* d_ws, size_t ws_size,
                              hipStream_t stream) {
    (void)in_sizes; (void)n_in; (void)out_size; (void)ws_size;
    const float*         plog = (const float*)d_in[0];          // [B,Q]
    const float*         lgv  = (const float*)d_in[1];          // [B,P,Q]
    const unsigned char* msk  = (const unsigned char*)d_in[2];  // [B,P,Q] bool
    const float*         segv = (const float*)d_in[3];          // [B,P,E]
    const float*         ppos = (const float*)d_in[4];          // [B,Q,2]
    const float*         chol = (const float*)d_in[5];          // [B,Q,2,2]
    const float*         tpos = (const float*)d_in[6];          // [B,E,2]
    const float*         isz  = (const float*)d_in[7];          // [B,2]
    float* out = (float*)d_out;
    float* W   = (float*)d_ws;

    // zero all accumulated regions (accP/accN/accM/colsums/nnz)
    hipMemsetAsync(W + ZERO_OFF, 0, (size_t)ZERO_LEN * sizeof(float), stream);

    k_rowstats<<<(Bc * Pc) / 8, 256, 0, stream>>>(lgv, msk, W + OFF_RMAX, W + OFF_RSUM);
    k_segstats<<<Bc * 16, 128, 0, stream>>>(segv, W + OFF_SEGC, W + OFF_NNZ);
    k_gemm3<<<Bc * QB * KS, 256, 0, stream>>>(lgv, msk, segv,
                                              W + OFF_RMAX, W + OFF_RSUM,
                                              W + OFF_ACCP, W + OFF_ACCN, W + OFF_ACCM,
                                              W + OFF_NEGC, W + OFF_PPC);
    long total = (long)Bc * Qc * Ec;
    k_combine<<<(unsigned)((total + 255) / 256), 256, 0, stream>>>(
        plog, ppos, chol, tpos, isz,
        W + OFF_ACCP, W + OFF_ACCN, W + OFF_ACCM,
        W + OFF_NEGC, W + OFF_PPC, W + OFF_SEGC, W + OFF_NNZ, out);
}
